// SusonoRoutedExperts_61753039781961
// MI455X (gfx1250) — compile-verified, matrix operand order checked by reading, not run on hardware
//
#include <hip/hip_runtime.h>

#define NUM_EXPERTS 8
#define HIDDEN 2048
#define INTER 1408
#define NTOK 8192
#define TILE_T 32
#define KCHUNK 128
#define MAXTILES (NTOK / TILE_T + NUM_EXPERTS)   // 264

// padded LDS strides (bf16 elements) to spread banks
#define XS (HIDDEN + 8)   // 2056
#define HS (INTER + 8)    // 1416
#define WSR (KCHUNK + 8)  // 136

typedef __attribute__((ext_vector_type(16))) __bf16 v16bf;
typedef __attribute__((ext_vector_type(8)))  __bf16 v8bf;
typedef __attribute__((ext_vector_type(8)))  float  v8f;

// ws layout (ints): [0..8) counts, [8..17) offsets, [24..32) cursors,
// [33] n_tiles, [64..64+2*MAXTILES) tile map {expert,row0}, [1024..1024+NTOK) sorted ids
#define WS_COUNTS 0
#define WS_OFFS   8
#define WS_CURS   24
#define WS_NTILES 33
#define WS_MAP    64
#define WS_SORTED 1024

__global__ void moe_init_kernel(int* __restrict__ ws) {
    int t = threadIdx.x;
    if (t < 64) ws[t] = 0;
}

__global__ void moe_count_kernel(const int* __restrict__ lids, int* __restrict__ ws) {
    int i = blockIdx.x * blockDim.x + threadIdx.x;
    if (i < NTOK) atomicAdd(&ws[WS_COUNTS + lids[i]], 1);
}

__global__ void moe_scan_kernel(int* __restrict__ ws) {
    if (threadIdx.x != 0 || blockIdx.x != 0) return;
    int off = 0, t = 0;
    for (int e = 0; e < NUM_EXPERTS; ++e) {
        int c = ws[WS_COUNTS + e];
        ws[WS_OFFS + e] = off;
        int nt = (c + TILE_T - 1) / TILE_T;
        for (int i = 0; i < nt; ++i) {
            ws[WS_MAP + 2 * t]     = e;
            ws[WS_MAP + 2 * t + 1] = off + i * TILE_T;
            ++t;
        }
        off += c;
    }
    ws[WS_OFFS + NUM_EXPERTS] = off;
    ws[WS_NTILES] = t;
}

__global__ void moe_scatter_kernel(const int* __restrict__ lids, int* __restrict__ ws) {
    int i = blockIdx.x * blockDim.x + threadIdx.x;
    if (i < NTOK) {
        int e = lids[i];
        int pos = ws[WS_OFFS + e] + atomicAdd(&ws[WS_CURS + e], 1);
        ws[WS_SORTED + pos] = i;
    }
}

// CDNA5 16-bit A/B fragment: lanes 0-15 take K = 0..7 and 16..23,
// lanes 16-31 take K = 8..15 and 24..31.  rowptr = &tile[row*stride + kbase].
__device__ __forceinline__ v16bf load_frag16(const __bf16* rowptr, int lane) {
    const int sel = (lane < 16) ? 0 : 8;
    const v8bf lo = *(const v8bf*)(rowptr + sel);
    const v8bf hi = *(const v8bf*)(rowptr + sel + 16);
    v16bf r;
#pragma unroll
    for (int i = 0; i < 8; ++i) { r[i] = lo[i]; r[i + 8] = hi[i]; }
    return r;
}

// Issue global loads for one 64x128 fp32 weight chunk into registers.
__device__ __forceinline__ void stage_regs(const float* __restrict__ src, int rowlen,
                                           int row0, int ksn, int tid, float4* rv) {
#pragma unroll
    for (int i = 0; i < 8; ++i) {
        int idx = tid + (i << 8);
        int r = idx >> 5;
        int c = (idx & 31) << 2;
        rv[i] = *(const float4*)(src + (size_t)(row0 + r) * rowlen + ksn + c);
    }
}

// Convert staged registers to bf16 and store into an LDS weight buffer.
__device__ __forceinline__ void store_stage(__bf16* __restrict__ dst, int tid, const float4* rv) {
#pragma unroll
    for (int i = 0; i < 8; ++i) {
        int idx = tid + (i << 8);
        int r = idx >> 5;
        int c = (idx & 31) << 2;
        __bf16* d = dst + r * WSR + c;
        d[0] = (__bf16)rv[i].x; d[1] = (__bf16)rv[i].y;
        d[2] = (__bf16)rv[i].z; d[3] = (__bf16)rv[i].w;
    }
}

__global__ __launch_bounds__(256) void moe_gemm_kernel(
    const float* __restrict__ tokens, const float* __restrict__ wts,
    const float* __restrict__ gup, const float* __restrict__ down,
    float* __restrict__ out, const int* __restrict__ ws)
{
    const int ntiles = ws[WS_NTILES];
    const int bid = blockIdx.x;
    if (bid >= ntiles) return;

    const int e    = ws[WS_MAP + 2 * bid];
    const int row0 = ws[WS_MAP + 2 * bid + 1];
    const int rend = ws[WS_OFFS + e + 1];

    const int tid  = threadIdx.x;
    const int lane = tid & 31;
    const int w    = tid >> 5;
    const int wr   = w & 1;    // token-half (0..1)
    const int wc   = w >> 1;   // 16-col group (0..3)

    extern __shared__ char smem[];
    __bf16* Xl  = (__bf16*)smem;             // [TILE_T][XS]
    __bf16* Hl  = Xl + TILE_T * XS;          // [TILE_T][HS]
    __bf16* Wg0 = Hl + TILE_T * HS;          // [64][WSR] x4 (double-buffered gate/up)
    __bf16* Wu0 = Wg0 + 64 * WSR;
    __bf16* Wg1 = Wu0 + 64 * WSR;
    __bf16* Wu1 = Wg1 + 64 * WSR;
    int*   idsS = (int*)(Wu1 + 64 * WSR);    // [TILE_T]
    float* wrS  = (float*)(idsS + TILE_T);   // [TILE_T]
    const int* sorted = ws + WS_SORTED;

    if (tid < TILE_T) {
        int rg = row0 + tid;
        if (rg >= rend) rg = rend - 1;       // clamp padded rows (tile exists => count>0)
        int tk = sorted[rg];
        idsS[tid] = tk;
        wrS[tid]  = wts[tk];
    }
    __syncthreads();

    // Stage token tile once: fp32 -> bf16, resident for all FC1 passes.
    for (int idx = tid; idx < TILE_T * (HIDDEN / 4); idx += 256) {
        int r = idx >> 9;                    // HIDDEN/4 == 512
        int c = (idx & 511) << 2;
        const float4 v = *(const float4*)(tokens + (size_t)idsS[r] * HIDDEN + c);
        __bf16* d = Xl + r * XS + c;
        d[0] = (__bf16)v.x; d[1] = (__bf16)v.y; d[2] = (__bf16)v.z; d[3] = (__bf16)v.w;
    }

    const float* Wge = gup  + (size_t)e * (2 * INTER) * HIDDEN;
    const float* Wde = down + (size_t)e * HIDDEN * INTER;
    const int mrow = wr * 16 + ((lane >= 16) ? 8 : 0);
    const int rowA = wr * 16 + (lane & 15);
    const int rowB = wc * 16 + (lane & 15);

    // ---------------- FC1 (gate & up) + SwiGLU -> Hl ----------------
    for (int nb = 0; nb < INTER; nb += 64) {
        v8f ag = {};
        v8f au = {};
        {   // prologue: stage chunk 0 into buffer 0
            float4 rg[8], ru[8];
            stage_regs(Wge, HIDDEN, nb, 0, tid, rg);
            stage_regs(Wge, HIDDEN, INTER + nb, 0, tid, ru);
            store_stage(Wg0, tid, rg);
            store_stage(Wu0, tid, ru);
        }
        __syncthreads();
        const __bf16* Xbase = Xl + (size_t)rowA * XS;
        int p = 0;
        for (int ks = 0; ks < HIDDEN; ks += KCHUNK, p ^= 1) {
            const __bf16* curG = (p ? Wg1 : Wg0) + (size_t)rowB * WSR;
            const __bf16* curU = (p ? Wu1 : Wu0) + (size_t)rowB * WSR;
            __bf16* nxtG = p ? Wg0 : Wg1;
            __bf16* nxtU = p ? Wu0 : Wu1;
            const bool hn = (ks + KCHUNK) < HIDDEN;
            float4 rg[8], ru[8];
            if (hn) {
                stage_regs(Wge, HIDDEN, nb, ks + KCHUNK, tid, rg);
                stage_regs(Wge, HIDDEN, INTER + nb, ks + KCHUNK, tid, ru);
                if (ks + 2 * KCHUNK < HIDDEN)   // warm L2 two chunks ahead
                    __builtin_prefetch(Wge + (size_t)(nb + (tid >> 3)) * HIDDEN +
                                       ks + 2 * KCHUNK + ((tid & 7) << 4), 0, 0);
            }
#pragma unroll
            for (int kk = 0; kk < KCHUNK; kk += 32) {
                v16bf a  = load_frag16(Xbase + ks + kk, lane);
                v16bf bg = load_frag16(curG + kk, lane);
                v16bf bu = load_frag16(curU + kk, lane);
                ag = __builtin_amdgcn_wmma_f32_16x16x32_bf16(false, a, false, bg, (short)0, ag, false, false);
                au = __builtin_amdgcn_wmma_f32_16x16x32_bf16(false, a, false, bu, (short)0, au, false, false);
            }
            if (hn) {
                store_stage(nxtG, tid, rg);
                store_stage(nxtU, tid, ru);
            }
            __syncthreads();
        }
        const int ncol = nb + wc * 16 + (lane & 15);
#pragma unroll
        for (int r = 0; r < 8; ++r) {
            float g = ag[r];
            float u = au[r];
            float h = (g / (1.0f + __expf(-g))) * u * wrS[mrow + r];
            Hl[(size_t)(mrow + r) * HS + ncol] = (__bf16)h;
        }
    }

    // ---------------- FC2: Hl @ down^T -> out ----------------
    for (int nb = 0; nb < HIDDEN; nb += 64) {
        v8f acc = {};
        {   // prologue: stage chunk 0 into buffer 0
            float4 rd[8];
            stage_regs(Wde, INTER, nb, 0, tid, rd);
            store_stage(Wg0, tid, rd);
        }
        __syncthreads();
        const __bf16* Hbase = Hl + (size_t)rowA * HS;
        int p = 0;
        for (int ks = 0; ks < INTER; ks += KCHUNK, p ^= 1) {   // 1408 = 11 * 128
            const __bf16* curB = (p ? Wg1 : Wg0) + (size_t)rowB * WSR;
            __bf16* nxtB = p ? Wg0 : Wg1;
            const bool hn = (ks + KCHUNK) < INTER;
            float4 rd[8];
            if (hn) stage_regs(Wde, INTER, nb, ks + KCHUNK, tid, rd);
#pragma unroll
            for (int kk = 0; kk < KCHUNK; kk += 32) {
                v16bf a = load_frag16(Hbase + ks + kk, lane);
                v16bf b = load_frag16(curB + kk, lane);
                acc = __builtin_amdgcn_wmma_f32_16x16x32_bf16(false, a, false, b, (short)0, acc, false, false);
            }
            if (hn) store_stage(nxtB, tid, rd);
            __syncthreads();
        }
        const int ncol = nb + wc * 16 + (lane & 15);
#pragma unroll
        for (int r = 0; r < 8; ++r) {
            int m = mrow + r;
            if (row0 + m < rend)
                out[(size_t)idsS[m] * HIDDEN + ncol] = acc[r];
        }
    }
}

static constexpr size_t SHMEM_BYTES =
    (size_t)(TILE_T * XS + TILE_T * HS + 4 * 64 * WSR) * 2 + TILE_T * 4 + TILE_T * 4;

extern "C" void kernel_launch(void* const* d_in, const int* in_sizes, int n_in,
                              void* d_out, int out_size, void* d_ws, size_t ws_size,
                              hipStream_t stream) {
    const float* tokens = (const float*)d_in[0];
    const int*   lids   = (const int*)d_in[1];
    const float* wts    = (const float*)d_in[2];
    const float* gup    = (const float*)d_in[3];
    const float* down   = (const float*)d_in[4];
    float* out = (float*)d_out;
    int*   ws  = (int*)d_ws;

    moe_init_kernel<<<1, 64, 0, stream>>>(ws);
    moe_count_kernel<<<NTOK / 256, 256, 0, stream>>>(lids, ws);
    moe_scan_kernel<<<1, 1, 0, stream>>>(ws);
    moe_scatter_kernel<<<NTOK / 256, 256, 0, stream>>>(lids, ws);
    moe_gemm_kernel<<<MAXTILES, 256, SHMEM_BYTES, stream>>>(tokens, wts, gup, down, out, ws);
}
